// DeepseekV3MoEToA2AAdapter_33122787787380
// MI455X (gfx1250) — compile-verified, matrix operand order checked by reading, not run on hardware
//
#include <hip/hip_runtime.h>
#include <hip/hip_bf16.h>

// ---------------------------------------------------------------------------
// DeepSeek-V3 MoE block for MI455X (gfx1250).
// Sparse top-2 routing + bf16 WMMA expert GEMMs (v_wmma_f32_16x16x32_bf16).
// A-tiles staged to LDS with global_load_async_to_lds_b128 (ASYNCcnt),
// B-operands streamed from L2-resident transposed bf16 weights (69 MB).
// ---------------------------------------------------------------------------

typedef __attribute__((ext_vector_type(16))) __bf16 v16bf;
typedef __attribute__((ext_vector_type(8)))  float  v8f;

constexpr int TB = 2 * 1024;   // tokens (B*S)
constexpr int HD = 1024;       // hidden
constexpr int NE = 8;          // experts
constexpr int ID = 1408;       // intermediate
constexpr int TK = 2;          // top-k
constexpr int NA = TB * TK;    // total assignments = 4096

constexpr int APAD_H = HD + 8;   // 1032 elems: row stride 516 dw ≡ 4 (mod 64 banks)
constexpr int APAD_I = ID + 8;   // 1416 elems: row stride 708 dw ≡ 4 (mod 64 banks)

__device__ __forceinline__ unsigned short f2bf(float f) {
  unsigned u = __float_as_uint(f);
  u += 0x7FFFu + ((u >> 16) & 1u);        // round-to-nearest-even
  return (unsigned short)(u >> 16);
}

union ABFrag { v16bf v; uint4 q[2]; };

__device__ __forceinline__ void async_load_b128(const void* gaddr, const void* lds_ptr) {
  unsigned loff = (unsigned)(unsigned long long)lds_ptr;  // low 32 bits = LDS offset
  asm volatile("global_load_async_to_lds_b128 %0, %1, off"
               :: "v"(loff), "v"(gaddr) : "memory");
}

__device__ __forceinline__ void wait_async0() {
#if __has_builtin(__builtin_amdgcn_s_wait_asynccnt)
  __builtin_amdgcn_s_wait_asynccnt(0);
#else
  asm volatile("s_wait_asynccnt 0x0" ::: "memory");
#endif
}

// ---------------------------------------------------------------------------
// 1) fp32 -> bf16 cast of activations x [T][H]
__global__ void k_cast_x(const float* __restrict__ x, unsigned short* __restrict__ xbf) {
  for (long long i = (long long)blockIdx.x * blockDim.x + threadIdx.x;
       i < (long long)TB * HD; i += (long long)gridDim.x * blockDim.x)
    xbf[i] = f2bf(x[i]);
}

// 2) gate/up [E][H][I] fp32 -> [E][I][H] bf16 (transpose: B-operand K-runs contiguous)
__global__ void k_tr_hi2ih(const float* __restrict__ g, const float* __restrict__ u,
                           unsigned short* __restrict__ gT, unsigned short* __restrict__ uT) {
  const long long N = (long long)NE * HD * ID;
  for (long long idx = (long long)blockIdx.x * blockDim.x + threadIdx.x;
       idx < N; idx += (long long)gridDim.x * blockDim.x) {
    int e = (int)(idx / ((long long)HD * ID));
    long long r = idx % ((long long)HD * ID);
    int h = (int)(r / ID), i = (int)(r % ID);
    long long o = ((long long)e * ID + i) * HD + h;
    gT[o] = f2bf(g[idx]);
    uT[o] = f2bf(u[idx]);
  }
}

// 3) down [E][I][H] fp32 -> [E][H][I] bf16
__global__ void k_tr_ih2hi(const float* __restrict__ d, unsigned short* __restrict__ dT) {
  const long long N = (long long)NE * ID * HD;
  for (long long idx = (long long)blockIdx.x * blockDim.x + threadIdx.x;
       idx < N; idx += (long long)gridDim.x * blockDim.x) {
    int e = (int)(idx / ((long long)ID * HD));
    long long r = idx % ((long long)ID * HD);
    int i = (int)(r / HD), h = (int)(r % HD);
    dT[((long long)e * HD + h) * ID + i] = f2bf(d[idx]);
  }
}

// ---------------------------------------------------------------------------
// 4) Router: logits -> softmax -> top-2 -> normalized weights + counts
__global__ void k_router(const float* __restrict__ x, const float* __restrict__ rw,
                         int* __restrict__ topk_e, float* __restrict__ topk_w,
                         int* __restrict__ cnt) {
  int t = blockIdx.x * blockDim.x + threadIdx.x;
  if (t >= TB) return;
  float acc[NE];
#pragma unroll
  for (int e = 0; e < NE; ++e) acc[e] = 0.f;
  const float* xr = x + (long long)t * HD;
  for (int h = 0; h < HD; ++h) {
    float xv = xr[h];
    const float4* w4 = (const float4*)(rw + (long long)h * NE);
    float4 wa = w4[0], wb = w4[1];
    acc[0] += xv * wa.x; acc[1] += xv * wa.y; acc[2] += xv * wa.z; acc[3] += xv * wa.w;
    acc[4] += xv * wb.x; acc[5] += xv * wb.y; acc[6] += xv * wb.z; acc[7] += xv * wb.w;
  }
  float mx = acc[0];
#pragma unroll
  for (int e = 1; e < NE; ++e) mx = fmaxf(mx, acc[e]);
  float p[NE], s = 0.f;
#pragma unroll
  for (int e = 0; e < NE; ++e) { p[e] = __expf(acc[e] - mx); s += p[e]; }
#pragma unroll
  for (int e = 0; e < NE; ++e) p[e] /= s;
  int e0 = 0; float p0 = p[0];
#pragma unroll
  for (int e = 1; e < NE; ++e) if (p[e] > p0) { p0 = p[e]; e0 = e; }
  int e1 = -1; float p1 = -1.f;
#pragma unroll
  for (int e = 0; e < NE; ++e) if (e != e0 && p[e] > p1) { p1 = p[e]; e1 = e; }
  float nrm = p0 + p1;
  topk_e[2 * t + 0] = e0; topk_w[2 * t + 0] = p0 / nrm;
  topk_e[2 * t + 1] = e1; topk_w[2 * t + 1] = p1 / nrm;
  atomicAdd(&cnt[e0], 1);
  atomicAdd(&cnt[e1], 1);
}

// 5) prefix-sum over 8 expert counts -> compact row space [0, 4096)
__global__ void k_prefix(const int* __restrict__ cnt, int* __restrict__ offs,
                         int* __restrict__ cnt2) {
  if (threadIdx.x == 0) {
    int o = 0;
    for (int e = 0; e < NE; ++e) { offs[e] = o; o += cnt[e]; cnt2[e] = 0; }
    offs[NE] = o;
  }
}

// 6) assign each (token, slot) a compact row; record token list per row
__global__ void k_assign(const int* __restrict__ topk_e, const int* __restrict__ offs,
                         int* __restrict__ cnt2, int* __restrict__ list_tok,
                         int* __restrict__ assign_row) {
  int t = blockIdx.x * blockDim.x + threadIdx.x;
  if (t >= TB) return;
#pragma unroll
  for (int s = 0; s < TK; ++s) {
    int e = topk_e[2 * t + s];
    int slot = atomicAdd(&cnt2[e], 1);
    int row = offs[e] + slot;
    list_tok[row] = t;
    assign_row[2 * t + s] = row;
  }
}

// ---------------------------------------------------------------------------
// 7) gate/up GEMM + SiLU*up.  Block = 256 threads (8 waves), tile M=32 x N=128.
//    Wave (mw,nw) owns M=16 x N=32 for BOTH gate and up: 4 accumulators sharing
//    one A fragment.  A staged in LDS via async DMA; B streamed from L2.
__global__ void k_gateup(const unsigned short* __restrict__ xbf,
                         const unsigned short* __restrict__ gT,   // [E][I][H]
                         const unsigned short* __restrict__ uT,   // [E][I][H]
                         const float* __restrict__ gbias,         // [E][I]
                         const float* __restrict__ ubias,         // [E][I]
                         const int* __restrict__ offs,
                         const int* __restrict__ list_tok,
                         unsigned short* __restrict__ act)        // [NA][I] bf16
{
  extern __shared__ unsigned short sA[];    // [32][APAD_H]

  const int e = blockIdx.z;
  const int base = offs[e];
  const int cntE = offs[e + 1] - base;
  const int mbase = blockIdx.y * 32;
  if (mbase >= cntE) return;

  const int tid = threadIdx.x;

  // ---- stage 32 gathered token rows (bf16, 2KB each) into LDS, async
  {
    int row = tid >> 3;                    // 0..31
    int sub = tid & 7;                     // 0..7
    int mrow = mbase + row; if (mrow >= cntE) mrow = cntE - 1;
    const int tok = list_tok[base + mrow];
    const unsigned short* src = xbf + (long long)tok * HD;
    unsigned short* dstrow = sA + row * APAD_H;
#pragma unroll
    for (int j = 0; j < 16; ++j) {
      int chunk = sub * 16 + j;            // 0..127 chunks of 8 elems
      async_load_b128(src + chunk * 8, dstrow + chunk * 8);
    }
  }
  wait_async0();
  __syncthreads();

  const int wave = tid >> 5;
  const int lane = tid & 31;
  const int m16  = lane & 15;
  const int hi   = lane >> 4;
  const int kb   = hi * 8;
  const int mw   = wave >> 2;              // 0..1  (M subtile)
  const int nw   = wave & 3;               // 0..3  (N subtile of 32)

  const unsigned short* arow = sA + (mw * 16 + m16) * APAD_H;

  int n[2];
  const unsigned short *bgp[2], *bup[2];
#pragma unroll
  for (int ss = 0; ss < 2; ++ss) {
    n[ss]  = blockIdx.x * 128 + nw * 32 + ss * 16 + m16;
    bgp[ss] = gT + ((long long)e * ID + n[ss]) * HD;
    bup[ss] = uT + ((long long)e * ID + n[ss]) * HD;
  }

  v8f accg[2] = {}, accu[2] = {};
#pragma unroll 2
  for (int k0 = 0; k0 < HD; k0 += 32) {
    ABFrag a;
    const unsigned short* a2 = arow + k0 + kb;
    a.q[0] = *(const uint4*)(a2);  a.q[1] = *(const uint4*)(a2 + 16);
#pragma unroll
    for (int ss = 0; ss < 2; ++ss) {
      ABFrag fg, fu;
      const unsigned short* g2 = bgp[ss] + k0 + kb;
      fg.q[0] = *(const uint4*)(g2); fg.q[1] = *(const uint4*)(g2 + 16);
      const unsigned short* u2 = bup[ss] + k0 + kb;
      fu.q[0] = *(const uint4*)(u2); fu.q[1] = *(const uint4*)(u2 + 16);
      accg[ss] = __builtin_amdgcn_wmma_f32_16x16x32_bf16(false, a.v, false, fg.v,
                                                         (short)0, accg[ss], false, false);
      accu[ss] = __builtin_amdgcn_wmma_f32_16x16x32_bf16(false, a.v, false, fu.v,
                                                         (short)0, accu[ss], false, false);
    }
  }

#pragma unroll
  for (int ss = 0; ss < 2; ++ss) {
    const float gb = gbias[(long long)e * ID + n[ss]];
    const float ub = ubias[(long long)e * ID + n[ss]];
#pragma unroll
    for (int j = 0; j < 8; ++j) {
      int m = mw * 16 + j + hi * 8;        // C/D layout: vgpr j -> M=j / M=8+j
      if (mbase + m < cntE) {
        float g = accg[ss][j] + gb;
        float u = accu[ss][j] + ub;
        float a = (g / (1.f + __expf(-g))) * u;   // SiLU(g) * u
        act[((long long)(base + mbase + m)) * ID + n[ss]] = f2bf(a);
      }
    }
  }
}

// 8) down GEMM (K=I=1408) + down_bias.  Block = 256 threads, tile M=32 x N=128;
//    wave owns M=16 x N=32 (2 accumulators).  A (act rows) staged async to LDS.
__global__ void k_down(const unsigned short* __restrict__ act,   // [NA][I] bf16
                       const unsigned short* __restrict__ dT,    // [E][H][I]
                       const float* __restrict__ dbias,          // [E][H]
                       const int* __restrict__ offs,
                       float* __restrict__ dws)                  // [NA][H] f32
{
  extern __shared__ unsigned short sA[];    // [32][APAD_I]

  const int e = blockIdx.z;
  const int base = offs[e];
  const int cntE = offs[e + 1] - base;
  const int mbase = blockIdx.y * 32;
  if (mbase >= cntE) return;

  const int tid = threadIdx.x;

  // ---- stage 32 act rows (88 chunks of 16B = first 704 of 1408 bf16) into LDS, async
  {
    int row = tid >> 3;
    int sub = tid & 7;
    int mrow = mbase + row; if (mrow >= cntE) mrow = cntE - 1;
    const unsigned short* src = act + (long long)(base + mrow) * ID;
    unsigned short* dstrow = sA + row * APAD_I;
#pragma unroll
    for (int j = 0; j < 11; ++j) {
      int chunk = sub * 11 + j;            // 0..87 chunks of 8 elems
      async_load_b128(src + chunk * 8, dstrow + chunk * 8);
    }
  }
  wait_async0();
  __syncthreads();

  const int wave = tid >> 5;
  const int lane = tid & 31;
  const int m16  = lane & 15;
  const int hi   = lane >> 4;
  const int kb   = hi * 8;
  const int mw   = wave >> 2;
  const int nw   = wave & 3;

  const unsigned short* arow = sA + (mw * 16 + m16) * APAD_I;

  int n[2];
  const unsigned short* bp[2];
#pragma unroll
  for (int ss = 0; ss < 2; ++ss) {
    n[ss] = blockIdx.x * 128 + nw * 32 + ss * 16 + m16;
    bp[ss] = dT + ((long long)e * HD + n[ss]) * ID;
  }

  v8f acc[2] = {};
#pragma unroll 2
  for (int k0 = 0; k0 < ID; k0 += 32) {
    ABFrag a;
    const unsigned short* a2 = arow + k0 + kb;
    a.q[0] = *(const uint4*)(a2);  a.q[1] = *(const uint4*)(a2 + 16);
#pragma unroll
    for (int ss = 0; ss < 2; ++ss) {
      ABFrag b;
      const unsigned short* b2 = bp[ss] + k0 + kb;
      b.q[0] = *(const uint4*)(b2); b.q[1] = *(const uint4*)(b2 + 16);
      acc[ss] = __builtin_amdgcn_wmma_f32_16x16x32_bf16(false, a.v, false, b.v,
                                                        (short)0, acc[ss], false, false);
    }
  }

#pragma unroll
  for (int ss = 0; ss < 2; ++ss) {
    const float db = dbias[(long long)e * HD + n[ss]];
#pragma unroll
    for (int j = 0; j < 8; ++j) {
      int m = mw * 16 + j + hi * 8;
      if (mbase + m < cntE)
        dws[((long long)(base + mbase + m)) * HD + n[ss]] = acc[ss][j] + db;
    }
  }
}

// 9) deterministic combine: y[t] = w0*d[row0] + w1*d[row1]
__global__ void k_combine(const float* __restrict__ dws, const int* __restrict__ assign_row,
                          const float* __restrict__ topk_w, float* __restrict__ y) {
  long long i = (long long)blockIdx.x * blockDim.x + threadIdx.x;
  if (i >= (long long)TB * HD) return;
  int t = (int)(i / HD), h = (int)(i % HD);
  int r0 = assign_row[2 * t + 0], r1 = assign_row[2 * t + 1];
  float w0 = topk_w[2 * t + 0],   w1 = topk_w[2 * t + 1];
  y[i] = w0 * dws[(long long)r0 * HD + h] + w1 * dws[(long long)r1 * HD + h];
}

// ---------------------------------------------------------------------------
extern "C" void kernel_launch(void* const* d_in, const int* in_sizes, int n_in,
                              void* d_out, int out_size, void* d_ws, size_t ws_size,
                              hipStream_t stream) {
  (void)in_sizes; (void)n_in; (void)out_size; (void)ws_size;
  const float* x     = (const float*)d_in[0];
  const float* rw    = (const float*)d_in[1];
  const float* gproj = (const float*)d_in[2];
  const float* uproj = (const float*)d_in[3];
  const float* dproj = (const float*)d_in[4];
  const float* gbias = (const float*)d_in[5];
  const float* ubias = (const float*)d_in[6];
  const float* dbias = (const float*)d_in[7];
  float* y = (float*)d_out;

  char* ws = (char*)d_ws;
  size_t off = 0;
  auto take = [&](size_t bytes) { size_t o = off; off = (off + bytes + 255) & ~(size_t)255; return o; };
  unsigned short* xbf  = (unsigned short*)(ws + take((size_t)TB * HD * 2));
  unsigned short* gT   = (unsigned short*)(ws + take((size_t)NE * ID * HD * 2));
  unsigned short* uT   = (unsigned short*)(ws + take((size_t)NE * ID * HD * 2));
  unsigned short* dT   = (unsigned short*)(ws + take((size_t)NE * HD * ID * 2));
  unsigned short* act  = (unsigned short*)(ws + take((size_t)NA * ID * 2));
  float*          dws  = (float*)         (ws + take((size_t)NA * HD * 4));
  int*   cnt        = (int*)  (ws + take(NE * 4));
  int*   offs       = (int*)  (ws + take((NE + 1) * 4));
  int*   cnt2       = (int*)  (ws + take(NE * 4));
  int*   topk_e     = (int*)  (ws + take(NA * 4));
  float* topk_w     = (float*)(ws + take(NA * 4));
  int*   list_tok   = (int*)  (ws + take(NA * 4));
  int*   assign_row = (int*)  (ws + take(NA * 4));

  hipMemsetAsync(cnt, 0, NE * 4, stream);

  { long long n = (long long)TB * HD;
    k_cast_x<<<(unsigned)((n + 255) / 256), 256, 0, stream>>>(x, xbf); }
  { long long n = (long long)NE * HD * ID;
    k_tr_hi2ih<<<(unsigned)((n + 255) / 256), 256, 0, stream>>>(gproj, uproj, gT, uT);
    k_tr_ih2hi<<<(unsigned)((n + 255) / 256), 256, 0, stream>>>(dproj, dT); }

  k_router<<<TB / 128, 128, 0, stream>>>(x, rw, topk_e, topk_w, cnt);
  k_prefix<<<1, 32, 0, stream>>>(cnt, offs, cnt2);
  k_assign<<<TB / 256, 256, 0, stream>>>(topk_e, offs, cnt2, list_tok, assign_row);

  // gate/up: N=1408 -> 11 blocks of 128 cols; up to 2048 tokens/expert -> 64 M-tiles
  k_gateup<<<dim3(ID / 128, TB / 32, NE), 256, 32 * APAD_H * 2, stream>>>(
      xbf, gT, uT, gbias, ubias, offs, list_tok, act);
  // down: N=1024 -> 8 blocks of 128 cols
  k_down<<<dim3(HD / 128, TB / 32, NE), 256, 32 * APAD_I * 2, stream>>>(
      act, dT, dbias, offs, dws);

  { long long n = (long long)TB * HD;
    k_combine<<<(unsigned)((n + 255) / 256), 256, 0, stream>>>(dws, assign_row, topk_w, y); }
}
